// P_SSSWA_87746181857949
// MI455X (gfx1250) — matrix operand order repaired and checked by reading, not compiled, not run on hardware
//
#include <hip/hip_runtime.h>

// ---------------------------------------------------------------------------
// MI455X (gfx1250, wave32) fused shifted-window attention.
// All GEMMs via v_wmma_f32_16x16x32_bf16; one workgroup (4 waves) per 8x8
// window. LDS operand layouts keep the WMMA k-index contiguous so every
// fragment is two 16B ds_load_b128 (A: kb, kb+16; B: kb, kb+16). Row stride
// padded to 72 bf16 (144B) for conflict-free wide LDS access.
// ---------------------------------------------------------------------------

typedef __attribute__((ext_vector_type(16))) __bf16 v16bf;
typedef __attribute__((ext_vector_type(8)))  __bf16 v8bf;
typedef __attribute__((ext_vector_type(8)))  float  v8f;

#define TPB 128   // 4 wave32s
#define LDP 72    // padded bf16 row stride (144 B)

__device__ inline v8f wmma_bf16(v16bf a, v16bf b, v8f c) {
  return __builtin_amdgcn_wmma_f32_16x16x32_bf16(false, a, false, b, (short)0, c,
                                                 false, false);
}

// A fragment (row-major [m][k], stride ld): lane L -> row m0+(L&15);
// K chunks {kb..kb+7} and {kb+16..kb+23}, kb = k0 + (L>=16 ? 8 : 0).
__device__ inline v16bf ld_a(const __bf16* base, int ld, int m0, int k0, int lane) {
  const __bf16* row = base + (size_t)(m0 + (lane & 15)) * ld;
  const int kb = k0 + ((lane >= 16) ? 8 : 0);
  v16bf r;
  *((v8bf*)&r)     = *(const v8bf*)(row + kb);
  *((v8bf*)&r + 1) = *(const v8bf*)(row + kb + 16);
  return r;
}

// B fragment from a single row pointer ([n][k] layout): K chunks {kb..kb+7}
// and {kb+16..kb+23}, kb = k0 + (L>=16 ? 8 : 0) (matches ld_a).
__device__ inline v16bf ld_b_row(const __bf16* row, int k0, int lane) {
  const int kb = k0 + ((lane >= 16) ? 8 : 0);
  v16bf r;
  *((v8bf*)&r)     = *(const v8bf*)(row + kb);
  *((v8bf*)&r + 1) = *(const v8bf*)(row + kb + 16);
  return r;
}

__device__ inline v16bf ld_b(const __bf16* base, int n0, int k0, int lane) {
  return ld_b_row(base + (size_t)(n0 + (lane & 15)) * LDP, k0, lane);
}

// C/D f32: lane L holds col n0+(L&15); VGPR r -> row m0 + r + (L>=16 ? 8 : 0)
template <typename F>
__device__ inline void store_d(v8f d, F f, int lane, int m0, int n0) {
  const int n  = n0 + (lane & 15);
  const int mb = m0 + ((lane >= 16) ? 8 : 0);
#pragma unroll
  for (int r = 0; r < 8; ++r) f(mb + r, n, d[r]);
}

// ---------------------------------------------------------------------------
// Weight repack: fp32 -> bf16, K padded 48->64 (zeros), proj convs tap-major.
// ---------------------------------------------------------------------------
__global__ void P_SSSWA_pack_weights(
    const float* __restrict__ w_qk, const float* __restrict__ w_qk_dw,
    const float* __restrict__ w_v,  const float* __restrict__ w_v_dw,
    const float* __restrict__ w_proj, const float* __restrict__ w_proj1,
    const float* __restrict__ w_qkv1, const float* __restrict__ w_qkv1_dw,
    __bf16* wqk_b, __bf16* wv_b, __bf16* wqkv_b,
    __bf16* wproj_b, __bf16* wproj1_b,
    float* dwqk, float* dwv, float* dwqkv) {
  const int i = blockIdx.x * blockDim.x + threadIdx.x;
  const int stride = gridDim.x * blockDim.x;
  for (int idx = i; idx < 96 * 64; idx += stride) {
    int m = idx >> 6, k = idx & 63;
    wqk_b[idx] = (k < 48) ? (__bf16)w_qk[m * 48 + k] : (__bf16)0.f;
  }
  for (int idx = i; idx < 48 * 64; idx += stride) {
    int m = idx >> 6, k = idx & 63;
    wv_b[idx] = (k < 48) ? (__bf16)w_v[m * 48 + k] : (__bf16)0.f;
  }
  for (int idx = i; idx < 144 * 64; idx += stride) {
    int m = idx >> 6, k = idx & 63;
    wqkv_b[idx] = (k < 48) ? (__bf16)w_qkv1[m * 48 + k] : (__bf16)0.f;
  }
  for (int idx = i; idx < 9 * 48 * 64; idx += stride) {
    int tap = idx / (48 * 64), rem = idx % (48 * 64);
    int o = rem >> 6, c = rem & 63;
    wproj_b[idx]  = (c < 48) ? (__bf16)w_proj [(o * 48 + c) * 9 + tap] : (__bf16)0.f;
    wproj1_b[idx] = (c < 48) ? (__bf16)w_proj1[(o * 48 + c) * 9 + tap] : (__bf16)0.f;
  }
  for (int idx = i; idx < 96 * 9;  idx += stride) dwqk[idx]  = w_qk_dw[idx];
  for (int idx = i; idx < 48 * 9;  idx += stride) dwv[idx]   = w_v_dw[idx];
  for (int idx = i; idx < 144 * 9; idx += stride) dwqkv[idx] = w_qkv1_dw[idx];
}

// ---------------------------------------------------------------------------
// Stage 1: plain-window cross attention (q,k from x_f; v from x) + proj conv.
// ---------------------------------------------------------------------------
__global__ __launch_bounds__(TPB) void P_SSSWA_stage1(
    const float* __restrict__ x, const float* __restrict__ xf,
    const __bf16* __restrict__ wqk, const __bf16* __restrict__ wv,
    const __bf16* __restrict__ wproj,
    const float* __restrict__ dwqk, const float* __restrict__ dwv,
    float* __restrict__ img) {
  __shared__ __align__(16) __bf16 sXf[64 * LDP];    // [pos][c]  B-op (k=c)
  __shared__ __align__(16) __bf16 sX [64 * LDP];    // [pos][c]  B-op
  __shared__ float  sQK0[96 * 64];                  // conv1x1 result
  __shared__ float  sQK1[96 * 64];                  // after depthwise 3x3
  __shared__ float  sV0[48 * 64];
  __shared__ float  sV1[48 * 64];
  __shared__ __align__(16) __bf16 sQt[64 * LDP];    // [i][c]    A-op (k=c)
  __shared__ __align__(16) __bf16 sKb[64 * LDP];    // [j][c]    B-op (k=c)
  __shared__ __align__(16) __bf16 sVb[64 * LDP];    // [c][h]    A-op (k=h)
  __shared__ float  sAttn[64 * 64];
  __shared__ __align__(16) __bf16 sAttnB[64 * LDP]; // [j][h]    B-op (k=h)
  __shared__ __align__(16) __bf16 sAO[64 * LDP];    // [p][c]    B-op (k=c)

  const int tid = threadIdx.x, lane = tid & 31, wid = tid >> 5;
  const int win = blockIdx.x;
  const int b = win >> 10, wrem = win & 1023;
  const int h0 = (wrem >> 5) * 8, w0 = (wrem & 31) * 8;

  // zero arrays whose k-dim is padded 48->64 (c)
  for (int i = tid; i < 64 * LDP; i += TPB) {
    __bf16 z = (__bf16)0.f;
    sXf[i] = z; sX[i] = z; sQt[i] = z; sKb[i] = z; sAO[i] = z;
  }
  __syncthreads();

  for (int i = tid; i < 48 * 64; i += TPB) {
    int c = i >> 6, p = i & 63;
    size_t g = (((size_t)(b * 48 + c) * 256) + (h0 + (p >> 3))) * 256 + (w0 + (p & 7));
    sX [p * LDP + c] = (__bf16)x[g];
    sXf[p * LDP + c] = (__bf16)xf[g];
  }
  __syncthreads();

  // conv1x1: qk = Wqk(96x48)@xf, v = Wv(48x48)@x ; nt = wave id
  {
    v16bf bq0 = ld_b(sXf, wid * 16, 0, lane), bq1 = ld_b(sXf, wid * 16, 32, lane);
    v16bf bv0 = ld_b(sX,  wid * 16, 0, lane), bv1 = ld_b(sX,  wid * 16, 32, lane);
#pragma unroll
    for (int mt = 0; mt < 6; ++mt) {
      v8f acc = {};
      acc = wmma_bf16(ld_a(wqk, 64, mt * 16, 0, lane),  bq0, acc);
      acc = wmma_bf16(ld_a(wqk, 64, mt * 16, 32, lane), bq1, acc);
      store_d(acc, [&](int m, int n, float v) { sQK0[m * 64 + n] = v; },
              lane, mt * 16, wid * 16);
    }
#pragma unroll
    for (int mt = 0; mt < 3; ++mt) {
      v8f acc = {};
      acc = wmma_bf16(ld_a(wv, 64, mt * 16, 0, lane),  bv0, acc);
      acc = wmma_bf16(ld_a(wv, 64, mt * 16, 32, lane), bv1, acc);
      store_d(acc, [&](int m, int n, float v) { sV0[m * 64 + n] = v; },
              lane, mt * 16, wid * 16);
    }
  }
  __syncthreads();

  // depthwise 3x3, pad=1 (per window, zero border)
  for (int i = tid; i < 96 * 64; i += TPB) {
    int ch = i >> 6, p = i & 63, r = p >> 3, c = p & 7;
    float s = 0.f;
#pragma unroll
    for (int dy = 0; dy < 3; ++dy)
#pragma unroll
      for (int dx = 0; dx < 3; ++dx) {
        int rr = r + dy - 1, cc = c + dx - 1;
        if (rr >= 0 && rr < 8 && cc >= 0 && cc < 8)
          s += dwqk[ch * 9 + dy * 3 + dx] * sQK0[ch * 64 + rr * 8 + cc];
      }
    sQK1[ch * 64 + p] = s;
  }
  for (int i = tid; i < 48 * 64; i += TPB) {
    int ch = i >> 6, p = i & 63, r = p >> 3, c = p & 7;
    float s = 0.f;
#pragma unroll
    for (int dy = 0; dy < 3; ++dy)
#pragma unroll
      for (int dx = 0; dx < 3; ++dx) {
        int rr = r + dy - 1, cc = c + dx - 1;
        if (rr >= 0 && rr < 8 && cc >= 0 && cc < 8)
          s += dwv[ch * 9 + dy * 3 + dx] * sV0[ch * 64 + rr * 8 + cc];
      }
    sV1[ch * 64 + p] = s;
    sVb[ch * LDP + p] = (__bf16)s;
  }
  __syncthreads();

  // L2 normalize q (ch 0..47 -> sQt[i][c]) and k (ch 48..95 -> sKb[j][c])
  if (tid < 96) {
    float ss = 0.f;
    for (int p = 0; p < 64; ++p) { float v = sQK1[tid * 64 + p]; ss += v * v; }
    float inv = 1.f / fmaxf(sqrtf(ss), 1e-12f);
    if (tid < 48)
      for (int p = 0; p < 64; ++p) sQt[p * LDP + tid] = (__bf16)(sQK1[tid * 64 + p] * inv);
    else
      for (int p = 0; p < 64; ++p) sKb[p * LDP + (tid - 48)] = (__bf16)(sQK1[tid * 64 + p] * inv);
  }
  __syncthreads();

  // attn = (Q^T K) / ws
  {
    v16bf bb0 = ld_b(sKb, wid * 16, 0, lane), bb1 = ld_b(sKb, wid * 16, 32, lane);
#pragma unroll
    for (int mt = 0; mt < 4; ++mt) {
      v8f acc = {};
      acc = wmma_bf16(ld_a(sQt, LDP, mt * 16, 0, lane),  bb0, acc);
      acc = wmma_bf16(ld_a(sQt, LDP, mt * 16, 32, lane), bb1, acc);
      store_d(acc, [&](int m, int n, float v) { sAttn[m * 64 + n] = v * 0.125f; },
              lane, mt * 16, wid * 16);
    }
  }
  __syncthreads();

  // row softmax over j; write transposed into sAttnB[j][h]
  if (tid < 64) {
    float mx = -1e30f;
    for (int j = 0; j < 64; ++j) mx = fmaxf(mx, sAttn[tid * 64 + j]);
    float s = 0.f;
    for (int j = 0; j < 64; ++j) s += __expf(sAttn[tid * 64 + j] - mx);
    float inv = 1.f / s;
    for (int j = 0; j < 64; ++j)
      sAttnB[j * LDP + tid] = (__bf16)(__expf(sAttn[tid * 64 + j] - mx) * inv);
  }
  __syncthreads();

  // out = V(48x64) @ attn(64x64) -> sAO[p][c]
  {
    v16bf bb0 = ld_b(sAttnB, wid * 16, 0, lane), bb1 = ld_b(sAttnB, wid * 16, 32, lane);
#pragma unroll
    for (int mt = 0; mt < 3; ++mt) {
      v8f acc = {};
      acc = wmma_bf16(ld_a(sVb, LDP, mt * 16, 0, lane),  bb0, acc);
      acc = wmma_bf16(ld_a(sVb, LDP, mt * 16, 32, lane), bb1, acc);
      store_d(acc, [&](int m, int n, float v) { sAO[n * LDP + m] = (__bf16)v; },
              lane, mt * 16, wid * 16);
    }
  }
  __syncthreads();

  // proj conv3x3: 9 tap-shifted GEMMs accumulated; shift = B row-select
  {
    v8f acc[3] = {v8f{}, v8f{}, v8f{}};
    const int n = wid * 16 + (lane & 15);
    for (int tap = 0; tap < 9; ++tap) {
      const int dy = tap / 3 - 1, dx = tap % 3 - 1;
      const int r = (n >> 3) + dy, c = (n & 7) + dx;
      const bool ok = (r >= 0 && r < 8 && c >= 0 && c < 8);
      const __bf16* row = sAO + (size_t)(ok ? (r * 8 + c) : 0) * LDP;
      v16bf z = {};
      v16bf bb0 = ld_b_row(row, 0, lane);  bb0 = ok ? bb0 : z;
      v16bf bb1 = ld_b_row(row, 32, lane); bb1 = ok ? bb1 : z;
      const __bf16* wt = wproj + tap * 48 * 64;
#pragma unroll
      for (int mt = 0; mt < 3; ++mt) {
        acc[mt] = wmma_bf16(ld_a(wt, 64, mt * 16, 0, lane),  bb0, acc[mt]);
        acc[mt] = wmma_bf16(ld_a(wt, 64, mt * 16, 32, lane), bb1, acc[mt]);
      }
    }
#pragma unroll
    for (int mt = 0; mt < 3; ++mt)
      store_d(acc[mt], [&](int m, int n2, float v) {
        size_t g = (((size_t)(b * 48 + m) * 256) + (h0 + (n2 >> 3))) * 256 + (w0 + (n2 & 7));
        img[g] = v;
      }, lane, mt * 16, wid * 16);
  }
}

// ---------------------------------------------------------------------------
// Stage 2: shifted-window self attention with analytic mask; roll fused into
// gather (+ss) and scatter (+ss).
// ---------------------------------------------------------------------------
__global__ __launch_bounds__(TPB) void P_SSSWA_stage2(
    const float* __restrict__ img,
    const __bf16* __restrict__ wqkv, const __bf16* __restrict__ wproj1,
    const float* __restrict__ dwqkv,
    float* __restrict__ out) {
  __shared__ __align__(16) __bf16 sX[64 * LDP];     // [pos][c]  B-op
  __shared__ float  sQ0[144 * 64];
  __shared__ float  sQ1[144 * 64];
  __shared__ __align__(16) __bf16 sQt[64 * LDP];    // [i][c]    A-op
  __shared__ __align__(16) __bf16 sKb[64 * LDP];    // [j][c]    B-op
  __shared__ __align__(16) __bf16 sVb[64 * LDP];    // [c][h]    A-op
  __shared__ float  sAttn[64 * 64];
  __shared__ __align__(16) __bf16 sAttnB[64 * LDP]; // [j][h]    B-op
  __shared__ __align__(16) __bf16 sAO[64 * LDP];    // [p][c]    B-op

  const int tid = threadIdx.x, lane = tid & 31, wid = tid >> 5;
  const int win = blockIdx.x;
  const int b = win >> 10, wrem = win & 1023;
  const int h0 = (wrem >> 5) * 8, w0 = (wrem & 31) * 8;

  for (int i = tid; i < 64 * LDP; i += TPB) {
    __bf16 z = (__bf16)0.f;
    sX[i] = z; sQt[i] = z; sKb[i] = z; sAO[i] = z;
  }
  __syncthreads();

  // gather shifted window: roll(out1, (-ss,-ss)) fused into index
  for (int i = tid; i < 48 * 64; i += TPB) {
    int c = i >> 6, p = i & 63;
    int hs = (h0 + (p >> 3) + 4) & 255, wsx = (w0 + (p & 7) + 4) & 255;
    sX[p * LDP + c] = (__bf16)img[(((size_t)(b * 48 + c) * 256) + hs) * 256 + wsx];
  }
  __syncthreads();

  // qkv conv1x1: 144x64
  {
    v16bf bb0 = ld_b(sX, wid * 16, 0, lane), bb1 = ld_b(sX, wid * 16, 32, lane);
#pragma unroll
    for (int mt = 0; mt < 9; ++mt) {
      v8f acc = {};
      acc = wmma_bf16(ld_a(wqkv, 64, mt * 16, 0, lane),  bb0, acc);
      acc = wmma_bf16(ld_a(wqkv, 64, mt * 16, 32, lane), bb1, acc);
      store_d(acc, [&](int m, int n, float v) { sQ0[m * 64 + n] = v; },
              lane, mt * 16, wid * 16);
    }
  }
  __syncthreads();

  for (int i = tid; i < 144 * 64; i += TPB) {
    int ch = i >> 6, p = i & 63, r = p >> 3, c = p & 7;
    float s = 0.f;
#pragma unroll
    for (int dy = 0; dy < 3; ++dy)
#pragma unroll
      for (int dx = 0; dx < 3; ++dx) {
        int rr = r + dy - 1, cc = c + dx - 1;
        if (rr >= 0 && rr < 8 && cc >= 0 && cc < 8)
          s += dwqkv[ch * 9 + dy * 3 + dx] * sQ0[ch * 64 + rr * 8 + cc];
      }
    sQ1[ch * 64 + p] = s;
    if (ch >= 96) sVb[(ch - 96) * LDP + p] = (__bf16)s;
  }
  __syncthreads();

  if (tid < 96) {
    float ss = 0.f;
    for (int p = 0; p < 64; ++p) { float v = sQ1[tid * 64 + p]; ss += v * v; }
    float inv = 1.f / fmaxf(sqrtf(ss), 1e-12f);
    if (tid < 48)
      for (int p = 0; p < 64; ++p) sQt[p * LDP + tid] = (__bf16)(sQ1[tid * 64 + p] * inv);
    else
      for (int p = 0; p < 64; ++p) sKb[p * LDP + (tid - 48)] = (__bf16)(sQ1[tid * 64 + p] * inv);
  }
  __syncthreads();

  // attn + shifted-window mask (analytic region ids in shifted coords)
  {
    v16bf bb0 = ld_b(sKb, wid * 16, 0, lane), bb1 = ld_b(sKb, wid * 16, 32, lane);
#pragma unroll
    for (int mt = 0; mt < 4; ++mt) {
      v8f acc = {};
      acc = wmma_bf16(ld_a(sQt, LDP, mt * 16, 0, lane),  bb0, acc);
      acc = wmma_bf16(ld_a(sQt, LDP, mt * 16, 32, lane), bb1, acc);
      store_d(acc, [&](int i2, int j2, float v) {
        int hi = h0 + (i2 >> 3), wi = w0 + (i2 & 7);
        int hj = h0 + (j2 >> 3), wj = w0 + (j2 & 7);
        int ri = ((hi < 248) ? 0 : (hi < 252) ? 1 : 2) * 3 +
                 ((wi < 248) ? 0 : (wi < 252) ? 1 : 2);
        int rj = ((hj < 248) ? 0 : (hj < 252) ? 1 : 2) * 3 +
                 ((wj < 248) ? 0 : (wj < 252) ? 1 : 2);
        sAttn[i2 * 64 + j2] = v * 0.125f + ((ri == rj) ? 0.f : -100.f);
      }, lane, mt * 16, wid * 16);
    }
  }
  __syncthreads();

  if (tid < 64) {
    float mx = -1e30f;
    for (int j = 0; j < 64; ++j) mx = fmaxf(mx, sAttn[tid * 64 + j]);
    float s = 0.f;
    for (int j = 0; j < 64; ++j) s += __expf(sAttn[tid * 64 + j] - mx);
    float inv = 1.f / s;
    for (int j = 0; j < 64; ++j)
      sAttnB[j * LDP + tid] = (__bf16)(__expf(sAttn[tid * 64 + j] - mx) * inv);
  }
  __syncthreads();

  {
    v16bf bb0 = ld_b(sAttnB, wid * 16, 0, lane), bb1 = ld_b(sAttnB, wid * 16, 32, lane);
#pragma unroll
    for (int mt = 0; mt < 3; ++mt) {
      v8f acc = {};
      acc = wmma_bf16(ld_a(sVb, LDP, mt * 16, 0, lane),  bb0, acc);
      acc = wmma_bf16(ld_a(sVb, LDP, mt * 16, 32, lane), bb1, acc);
      store_d(acc, [&](int m, int n, float v) { sAO[n * LDP + m] = (__bf16)v; },
              lane, mt * 16, wid * 16);
    }
  }
  __syncthreads();

  // proj1 conv3x3 + roll-back (+ss) fused into the output scatter
  {
    v8f acc[3] = {v8f{}, v8f{}, v8f{}};
    const int n = wid * 16 + (lane & 15);
    for (int tap = 0; tap < 9; ++tap) {
      const int dy = tap / 3 - 1, dx = tap % 3 - 1;
      const int r = (n >> 3) + dy, c = (n & 7) + dx;
      const bool ok = (r >= 0 && r < 8 && c >= 0 && c < 8);
      const __bf16* row = sAO + (size_t)(ok ? (r * 8 + c) : 0) * LDP;
      v16bf z = {};
      v16bf bb0 = ld_b_row(row, 0, lane);  bb0 = ok ? bb0 : z;
      v16bf bb1 = ld_b_row(row, 32, lane); bb1 = ok ? bb1 : z;
      const __bf16* wt = wproj1 + tap * 48 * 64;
#pragma unroll
      for (int mt = 0; mt < 3; ++mt) {
        acc[mt] = wmma_bf16(ld_a(wt, 64, mt * 16, 0, lane),  bb0, acc[mt]);
        acc[mt] = wmma_bf16(ld_a(wt, 64, mt * 16, 32, lane), bb1, acc[mt]);
      }
    }
#pragma unroll
    for (int mt = 0; mt < 3; ++mt)
      store_d(acc[mt], [&](int m, int n2, float v) {
        int hg = (h0 + (n2 >> 3) + 4) & 255, wg = (w0 + (n2 & 7) + 4) & 255;
        out[(((size_t)(b * 48 + m) * 256) + hg) * 256 + wg] = v;
      }, lane, mt * 16, wid * 16);
  }
}

// ---------------------------------------------------------------------------
extern "C" void kernel_launch(void* const* d_in, const int* in_sizes, int n_in,
                              void* d_out, int out_size, void* d_ws, size_t ws_size,
                              hipStream_t stream) {
  const float* x         = (const float*)d_in[0];
  const float* x_f       = (const float*)d_in[1];
  const float* w_qk      = (const float*)d_in[2];
  const float* w_qk_dw   = (const float*)d_in[3];
  const float* w_v       = (const float*)d_in[4];
  const float* w_v_dw    = (const float*)d_in[5];
  const float* w_proj    = (const float*)d_in[6];
  const float* w_proj1   = (const float*)d_in[7];
  const float* w_qkv1    = (const float*)d_in[8];
  const float* w_qkv1_dw = (const float*)d_in[9];
  float* out = (float*)d_out;

  char* ws = (char*)d_ws;
  const size_t IMG_BYTES = (size_t)8 * 48 * 256 * 256 * 4;  // stage-1 image
  float*  img      = (float*)ws;
  __bf16* wqk_b    = (__bf16*)(ws + IMG_BYTES);
  __bf16* wv_b     = wqk_b   + 96 * 64;
  __bf16* wqkv_b   = wv_b    + 48 * 64;
  __bf16* wproj_b  = wqkv_b  + 144 * 64;
  __bf16* wproj1_b = wproj_b + 9 * 48 * 64;
  float*  dwqk     = (float*)(wproj1_b + 9 * 48 * 64);
  float*  dwv      = dwqk + 96 * 9;
  float*  dwqkv    = dwv  + 48 * 9;

  P_SSSWA_pack_weights<<<64, 256, 0, stream>>>(
      w_qk, w_qk_dw, w_v, w_v_dw, w_proj, w_proj1, w_qkv1, w_qkv1_dw,
      wqk_b, wv_b, wqkv_b, wproj_b, wproj1_b, dwqk, dwv, dwqkv);

  P_SSSWA_stage1<<<8192, TPB, 0, stream>>>(
      x, x_f, wqk_b, wv_b, wproj_b, dwqk, dwv, img);

  P_SSSWA_stage2<<<8192, TPB, 0, stream>>>(
      img, wqkv_b, wproj1_b, dwqkv, out);
}
